// FixedSOTATabularAutoencoder_34428457845297
// MI455X (gfx1250) — compile-verified, matrix operand order checked by reading, not run on hardware
//
#include <hip/hip_runtime.h>
#include <math.h>

typedef __attribute__((ext_vector_type(16))) _Float16 v16h;
typedef __attribute__((ext_vector_type(8)))  _Float16 v8h;
typedef __attribute__((ext_vector_type(8)))  float    v8f;
typedef _Float16 h16;

#define BM 128
#define BN 64
#define BK 32

// ---------------------------------------------------------------- helpers
__device__ __forceinline__ float gelu_exact(float x) {
    return 0.5f * x * (1.0f + erff(x * 0.70710678118654752f));
}

// Load one 16x32 f16 fragment from LDS (row-major, row stride = BK halfs).
// ISA layout (16-bit A / B^T): lane m = lane&15, kb = (lane>>4)*8;
// elements 0..7 = K kb..kb+7 ; elements 8..15 = K kb+16..kb+23.
__device__ __forceinline__ v16h load_frag(const h16* __restrict__ base) {
    const int lane = threadIdx.x & 31;
    const int m  = lane & 15;
    const int kb = (lane >> 4) << 3;
    const h16* p = base + m * BK + kb;
    v8h lo = *(const v8h*)(p);
    v8h hi = *(const v8h*)(p + 16);
    return __builtin_shufflevector(lo, hi, 0,1,2,3,4,5,6,7,8,9,10,11,12,13,14,15);
}

// ---------------------------------------------------------------- WMMA GEMM
// C[M,N] = A[M,K](f16) @ B[K,N](f16) (+ bias), B supplied TRANSPOSED: Bt[N][K], ldb >= K.
// OUTMODE 0: f32 out, 1: f16 out ; RESID: Cf += acc + bias
template<int OUTMODE, bool RESID>
__global__ __launch_bounds__(256) void gemm_wmma(
    const h16* __restrict__ A, int lda,
    const h16* __restrict__ Bt, int ldb,
    const float* __restrict__ bias,
    float* __restrict__ Cf, h16* __restrict__ Ch, int ldc, int K)
{
    __shared__ __align__(16) h16 As[BM * BK];
    __shared__ __align__(16) h16 Bs[BN * BK];   // [n][k]

    const int m0 = blockIdx.y * BM;
    const int n0 = blockIdx.x * BN;
    const unsigned tid  = threadIdx.x;
    const unsigned lane = tid & 31;
    const unsigned wave = tid >> 5;
    const int wm = (wave & 3) * 32;
    const int wn = (wave >> 2) * 32;

    v8f acc[2][2] = {};

    for (int k0 = 0; k0 < K; k0 += BK) {
        {   // A tile: 128x32 halfs, 16 per thread (two v8h)
            const int r = tid >> 1;
            const int cc = (tid & 1) * 16;
            const h16* gp = A + (size_t)(m0 + r) * lda + k0 + cc;
            *(v8h*)(&As[r * BK + cc])     = *(const v8h*)(gp);
            *(v8h*)(&As[r * BK + cc + 8]) = *(const v8h*)(gp + 8);
            if (k0 + BK < K) __builtin_prefetch((const void*)(gp + BK), 0, 1);
        }
        {   // B^T tile: 64x32 halfs, 8 per thread (one v8h), fully contiguous
            const int r = tid >> 2;
            const int cc = (tid & 3) * 8;
            const h16* gp = Bt + (size_t)(n0 + r) * ldb + k0 + cc;
            *(v8h*)(&Bs[r * BK + cc]) = *(const v8h*)(gp);
        }
        __syncthreads();
        v16h af0 = load_frag(&As[(wm +  0) * BK]);
        v16h af1 = load_frag(&As[(wm + 16) * BK]);
        v16h bf0 = load_frag(&Bs[(wn +  0) * BK]);
        v16h bf1 = load_frag(&Bs[(wn + 16) * BK]);
        acc[0][0] = __builtin_amdgcn_wmma_f32_16x16x32_f16(false, af0, false, bf0, (short)0, acc[0][0], false, false);
        acc[0][1] = __builtin_amdgcn_wmma_f32_16x16x32_f16(false, af0, false, bf1, (short)0, acc[0][1], false, false);
        acc[1][0] = __builtin_amdgcn_wmma_f32_16x16x32_f16(false, af1, false, bf0, (short)0, acc[1][0], false, false);
        acc[1][1] = __builtin_amdgcn_wmma_f32_16x16x32_f16(false, af1, false, bf1, (short)0, acc[1][1], false, false);
        __syncthreads();
    }

    const int cn    = lane & 15;
    const int rbase = (lane >> 4) * 8;
    #pragma unroll
    for (int i = 0; i < 2; ++i) {
        #pragma unroll
        for (int j = 0; j < 2; ++j) {
            const int col = n0 + wn + j * 16 + cn;
            const float bv = bias ? bias[col] : 0.0f;
            #pragma unroll
            for (int r = 0; r < 8; ++r) {
                const int row = m0 + wm + i * 16 + rbase + r;
                const float val = acc[i][j][r] + bv;
                if (RESID)              Cf[(size_t)row * ldc + col] += val;
                else if (OUTMODE == 1)  Ch[(size_t)row * ldc + col] = (h16)val;
                else                    Cf[(size_t)row * ldc + col] = val;
            }
        }
    }
}

// Fused FFN1: act = gelu(A@Wu + bu) * (A@Wg + bg)
// WuT = rows [0,1024) of transposed ffn_w1 (each row K=256), WgT = WuT + 1024*K.
__global__ __launch_bounds__(256) void gemm_ffn1(
    const h16* __restrict__ A, int lda,
    const h16* __restrict__ WuT, const h16* __restrict__ WgT, int ldb,
    const float* __restrict__ bias,              // ffn_b1 (2048)
    h16* __restrict__ Ch, int ldc, int K)
{
    __shared__ __align__(16) h16 As[BM * BK];
    __shared__ __align__(16) h16 Bu[BN * BK];
    __shared__ __align__(16) h16 Bg[BN * BK];

    const int m0 = blockIdx.y * BM;
    const int n0 = blockIdx.x * BN;                // in [0,1024)
    const unsigned tid  = threadIdx.x;
    const unsigned lane = tid & 31;
    const unsigned wave = tid >> 5;
    const int wm = (wave & 3) * 32;
    const int wn = (wave >> 2) * 32;

    v8f au[2][2] = {};
    v8f ag[2][2] = {};

    for (int k0 = 0; k0 < K; k0 += BK) {
        {
            const int r = tid >> 1;
            const int cc = (tid & 1) * 16;
            const h16* gp = A + (size_t)(m0 + r) * lda + k0 + cc;
            *(v8h*)(&As[r * BK + cc])     = *(const v8h*)(gp);
            *(v8h*)(&As[r * BK + cc + 8]) = *(const v8h*)(gp + 8);
        }
        {
            const int r = tid >> 2;
            const int cc = (tid & 3) * 8;
            const size_t go = (size_t)(n0 + r) * ldb + k0 + cc;
            *(v8h*)(&Bu[r * BK + cc]) = *(const v8h*)(WuT + go);
            *(v8h*)(&Bg[r * BK + cc]) = *(const v8h*)(WgT + go);
        }
        __syncthreads();
        v16h af0 = load_frag(&As[(wm +  0) * BK]);
        v16h af1 = load_frag(&As[(wm + 16) * BK]);
        v16h bu0 = load_frag(&Bu[(wn +  0) * BK]);
        v16h bu1 = load_frag(&Bu[(wn + 16) * BK]);
        v16h bg0 = load_frag(&Bg[(wn +  0) * BK]);
        v16h bg1 = load_frag(&Bg[(wn + 16) * BK]);
        au[0][0] = __builtin_amdgcn_wmma_f32_16x16x32_f16(false, af0, false, bu0, (short)0, au[0][0], false, false);
        au[0][1] = __builtin_amdgcn_wmma_f32_16x16x32_f16(false, af0, false, bu1, (short)0, au[0][1], false, false);
        au[1][0] = __builtin_amdgcn_wmma_f32_16x16x32_f16(false, af1, false, bu0, (short)0, au[1][0], false, false);
        au[1][1] = __builtin_amdgcn_wmma_f32_16x16x32_f16(false, af1, false, bu1, (short)0, au[1][1], false, false);
        ag[0][0] = __builtin_amdgcn_wmma_f32_16x16x32_f16(false, af0, false, bg0, (short)0, ag[0][0], false, false);
        ag[0][1] = __builtin_amdgcn_wmma_f32_16x16x32_f16(false, af0, false, bg1, (short)0, ag[0][1], false, false);
        ag[1][0] = __builtin_amdgcn_wmma_f32_16x16x32_f16(false, af1, false, bg0, (short)0, ag[1][0], false, false);
        ag[1][1] = __builtin_amdgcn_wmma_f32_16x16x32_f16(false, af1, false, bg1, (short)0, ag[1][1], false, false);
        __syncthreads();
    }

    const int cn    = lane & 15;
    const int rbase = (lane >> 4) * 8;
    #pragma unroll
    for (int i = 0; i < 2; ++i) {
        #pragma unroll
        for (int j = 0; j < 2; ++j) {
            const int col = n0 + wn + j * 16 + cn;
            const float bu_ = bias[col];
            const float bg_ = bias[1024 + col];
            #pragma unroll
            for (int r = 0; r < 8; ++r) {
                const int row = m0 + wm + i * 16 + rbase + r;
                const float u = au[i][j][r] + bu_;
                const float g = ag[i][j][r] + bg_;
                Ch[(size_t)row * ldc + col] = (h16)(gelu_exact(u) * g);
            }
        }
    }
}

// ---------------------------------------------------------------- small kernels
// convert f32 [K][N] -> f16 transposed [N][K]
__global__ void cvt_t_kernel(const float* __restrict__ s, h16* __restrict__ d, int K, int N) {
    int i = blockIdx.x * 256 + threadIdx.x;
    if (i < K * N) {
        int k = i / N, n = i - k * N;
        d[(size_t)n * K + k] = (h16)s[i];
    }
}

__global__ __launch_bounds__(64) void xn_kernel(
    const float* __restrict__ x, const float* __restrict__ fm,
    const float* __restrict__ fs, float* __restrict__ xn)
{
    int b = blockIdx.x, f = threadIdx.x;
    xn[b * 64 + f] = (x[b * 64 + f] - fm[f]) / (fs[f] + 1e-8f);
}

__global__ __launch_bounds__(32) void edges_kernel(
    const float* __restrict__ bin_logits, float* __restrict__ edges)
{
    int f = blockIdx.x, t = threadIdx.x;
    __shared__ float e[32];
    float x = bin_logits[f * 32 + t];
    e[t] = (x > 20.f) ? x : log1pf(__expf(x));
    __syncthreads();
    if (t == 0) {
        float c = 0.f;
        for (int i = 0; i < 32; ++i) { c += e[i]; e[i] = c; }
        float mu = 0.f;
        for (int i = 0; i < 32; ++i) mu += e[i];
        mu *= (1.f / 32.f);
        float var = 0.f;
        for (int i = 0; i < 32; ++i) { float d = e[i] - mu; var += d * d; }
        var *= (1.f / 31.f);
        float inv = 4.f * rsqrtf(var);
        float* out = edges + f * 34;
        for (int i = 0; i < 32; ++i) out[1 + i] = (e[i] - mu) * inv;
        out[0]  = out[1]  - 1.f;
        out[33] = out[32] + 1.f;
    }
}

__global__ __launch_bounds__(64) void odt_kernel(
    const float* __restrict__ xn, const float* __restrict__ sel_w,
    const float* __restrict__ sel_b, const float* __restrict__ leaf_values,
    float* __restrict__ odt)
{
    int b = blockIdx.x, t = threadIdx.x;
    __shared__ float xr[64], p[24], path[64];
    xr[t] = xn[b * 64 + t];
    __syncthreads();
    if (t < 24) {
        int lvl = t / 8, tr = t % 8;
        float s = sel_b[lvl * 8 + tr];
        for (int f = 0; f < 64; ++f) s += xr[f] * sel_w[(lvl * 64 + f) * 8 + tr];
        p[t] = 1.f / (1.f + __expf(-s));
    }
    __syncthreads();
    {
        int tr = t >> 3, leaf = t & 7;
        float pr = 1.f;
        for (int lvl = 0; lvl < 3; ++lvl) {
            int bit = (leaf >> (2 - lvl)) & 1;
            float pv = p[lvl * 8 + tr];
            pr *= (bit == 0) ? pv : (1.f - pv);
        }
        path[t] = pr;
    }
    __syncthreads();
    float acc = 0.f;
    for (int tr = 0; tr < 8; ++tr)
        for (int lf = 0; lf < 8; ++lf)
            acc += path[tr * 8 + lf] * leaf_values[((size_t)tr * 8 + lf) * 64 + t];
    odt[b * 64 + t] = acc * 0.125f;
}

// per (b,f): gelu feature MLP -> tok(128) ; assemble comb16 row [tok|pos|bine|odt] (320)
__global__ __launch_bounds__(128) void tok_kernel(
    const float* __restrict__ xn, const float* __restrict__ fp_w1,
    const float* __restrict__ fp_b1, const float* __restrict__ fp_w2,
    const float* __restrict__ fp_b2, const float* __restrict__ pos_emb,
    const float* __restrict__ edges, const float* __restrict__ bin_emb_w,
    const float* __restrict__ odt, h16* __restrict__ comb16)
{
    int bf = blockIdx.x, b = bf >> 6, f = bf & 63, t = threadIdx.x;
    __shared__ float h1[64];
    __shared__ int binS;
    float xv = xn[bf];
    if (t < 64) h1[t] = gelu_exact(xv * fp_w1[f * 64 + t] + fp_b1[f * 64 + t]);
    if (t == 0) {
        const float* e = edges + f * 34;
        int idx = 34;
        for (int i = 0; i < 34; ++i) { if (e[i] >= xv) { idx = i; break; } }
        int bin = idx - 1;
        binS = bin < 0 ? 0 : (bin > 31 ? 31 : bin);
    }
    __syncthreads();
    float acc = fp_b2[f * 128 + t];
    for (int j = 0; j < 64; ++j) acc += h1[j] * fp_w2[((size_t)f * 64 + j) * 128 + t];
    h16* crow = comb16 + (size_t)bf * 320;
    crow[t] = (h16)acc;
    if (t < 64) {
        crow[128 + t] = (h16)pos_emb[f * 64 + t];
        crow[192 + t] = (h16)bin_emb_w[binS * 64 + t];
        crow[256 + t] = (h16)odt[b * 64 + t];
    }
}

__global__ __launch_bounds__(256) void assemble_kernel(
    const float* __restrict__ tmp, const float* __restrict__ cls_token,
    float* __restrict__ h)
{
    int r = blockIdx.x, t = threadIdx.x;
    int b = r / 65, l = r % 65;
    h[(size_t)r * 256 + t] = (l == 0) ? cls_token[t]
                                      : tmp[((size_t)b * 64 + (l - 1)) * 256 + t];
}

// adaptive norm: softmax(gate)-mixed LN/RMS/BN, f16 out (ld 256)
__global__ __launch_bounds__(256) void anorm_kernel(
    const float* __restrict__ in, long long row_stride,
    const float* __restrict__ ln_w, const float* __restrict__ ln_b,
    const float* __restrict__ rms_w, const float* __restrict__ bn_g,
    const float* __restrict__ bn_b, const float* __restrict__ gate,
    int idx, h16* __restrict__ out)
{
    const int row = blockIdx.x, d = threadIdx.x;
    const float hv = in[(size_t)row * row_stride + d];
    __shared__ float rS[8], rS2[8];
    __shared__ float muS, varS, msS;
    float s = hv, s2 = hv * hv;
    #pragma unroll
    for (int o = 16; o > 0; o >>= 1) {
        s  += __shfl_down(s,  o, 32);
        s2 += __shfl_down(s2, o, 32);
    }
    if ((d & 31) == 0) { rS[d >> 5] = s; rS2[d >> 5] = s2; }
    __syncthreads();
    if (d == 0) {
        float ts = 0.f, ts2 = 0.f;
        for (int i = 0; i < 8; ++i) { ts += rS[i]; ts2 += rS2[i]; }
        float mu = ts * (1.f / 256.f);
        muS = mu; msS = ts2 * (1.f / 256.f); varS = msS - mu * mu;
    }
    __syncthreads();
    float g0 = gate[idx * 3 + 0], g1 = gate[idx * 3 + 1], g2 = gate[idx * 3 + 2];
    float mg = fmaxf(g0, fmaxf(g1, g2));
    float e0 = __expf(g0 - mg), e1 = __expf(g1 - mg), e2 = __expf(g2 - mg);
    float inv = 1.f / (e0 + e1 + e2);
    float ln = (hv - muS) * rsqrtf(varS + 1e-5f) * ln_w[idx * 256 + d] + ln_b[idx * 256 + d];
    float rm = rms_w[idx * 256 + d] * hv * rsqrtf(msS + 1e-6f);
    float bn = hv * rsqrtf(1.f + 1e-5f) * bn_g[idx * 256 + d] + bn_b[idx * 256 + d];
    out[(size_t)row * 256 + d] = (h16)(e0 * inv * ln + e1 * inv * rm + e2 * inv * bn);
}

// attention probs: softmax(q k^T / sqrt(32)) per (b,h)
__global__ __launch_bounds__(128) void attn_kernel(
    const h16* __restrict__ q16, const h16* __restrict__ k16,
    float* __restrict__ attnP)
{
    const int bh = blockIdx.x, b = bh >> 3, hh = bh & 7;
    const int t = threadIdx.x;
    __shared__ float qv[32];
    __shared__ float red[128];
    const float scale = 0.1767766952966369f;   // 1/sqrt(32)
    for (int qr = 0; qr < 65; ++qr) {
        if (t < 32) qv[t] = (float)q16[((size_t)(b * 65 + qr)) * 256 + hh * 32 + t];
        __syncthreads();
        float sc = -1e30f;
        if (t < 65) {
            float s = 0.f;
            const h16* kp = &k16[((size_t)(b * 65 + t)) * 256 + hh * 32];
            for (int d = 0; d < 32; ++d) s += qv[d] * (float)kp[d];
            sc = s * scale;
        }
        red[t] = sc; __syncthreads();
        for (int o = 64; o > 0; o >>= 1) { if (t < o) red[t] = fmaxf(red[t], red[t + o]); __syncthreads(); }
        float mx = red[0]; __syncthreads();
        float e = (t < 65) ? __expf(sc - mx) : 0.f;
        red[t] = e; __syncthreads();
        for (int o = 64; o > 0; o >>= 1) { if (t < o) red[t] += red[t + o]; __syncthreads(); }
        float sm = red[0]; __syncthreads();
        if (t < 65) attnP[(((size_t)bh) * 65 + qr) * 65 + t] = e / sm;
        __syncthreads();
    }
}

// grouped conv over sequence: vc[b,l,c] = sum_i sum_j v[b,l+i-pad,g*32+j]*w[c,j,i] + bias[c]
__global__ __launch_bounds__(256) void gconv_kernel(
    const float* __restrict__ v, const float* __restrict__ cw,
    const float* __restrict__ cb, int ks, float* __restrict__ vc)
{
    const int b = blockIdx.x, c = threadIdx.x, g = c >> 5;
    const int pad = ks >> 1;
    const float* wbase = cw + (size_t)c * 32 * ks;
    for (int l = 0; l < 65; ++l) {
        float acc = cb[c];
        for (int i = 0; i < ks; ++i) {
            int tt = l + i - pad;
            if (tt < 0 || tt >= 65) continue;
            const float* vrow = v + ((size_t)(b * 65 + tt)) * 256 + g * 32;
            for (int j = 0; j < 32; ++j) acc += vrow[j] * wbase[j * ks + i];
        }
        vc[((size_t)(b * 65 + l)) * 256 + c] = acc;
    }
}

// o_s = attn @ vc -> outs16[:, col0 + c] (f16, ld 768)
__global__ __launch_bounds__(256) void av_kernel(
    const float* __restrict__ attnP, const float* __restrict__ vc,
    h16* __restrict__ outs16, int col0)
{
    const int b = blockIdx.x, c = threadIdx.x, g = c >> 5;
    __shared__ float arow[520];    // [8 heads][65]
    for (int l = 0; l < 65; ++l) {
        for (int idx = c; idx < 520; idx += 256) {
            int hh = idx / 65, tt = idx % 65;
            arow[idx] = attnP[(((size_t)(b * 8 + hh)) * 65 + l) * 65 + tt];
        }
        __syncthreads();
        float acc = 0.f;
        for (int t = 0; t < 65; ++t)
            acc += arow[g * 65 + t] * vc[((size_t)(b * 65 + t)) * 256 + c];
        outs16[((size_t)(b * 65 + l)) * 768 + col0 + c] = (h16)acc;
        __syncthreads();
    }
}

// ---------------------------------------------------------------- host
extern "C" void kernel_launch(void* const* d_in, const int* in_sizes, int n_in,
                              void* d_out, int out_size, void* d_ws, size_t ws_size,
                              hipStream_t stream)
{
    const float* x          = (const float*)d_in[0];
    const float* fmean      = (const float*)d_in[1];
    const float* fscale     = (const float*)d_in[2];
    const float* fp_w1      = (const float*)d_in[3];
    const float* fp_b1      = (const float*)d_in[4];
    const float* fp_w2      = (const float*)d_in[5];
    const float* fp_b2      = (const float*)d_in[6];
    const float* pos_emb    = (const float*)d_in[7];
    const float* bin_logits = (const float*)d_in[8];
    const float* bin_emb_w  = (const float*)d_in[9];
    const float* sel_w      = (const float*)d_in[10];
    const float* sel_b      = (const float*)d_in[11];
    const float* leaf_values= (const float*)d_in[12];
    const float* proj_w     = (const float*)d_in[13];
    const float* proj_b     = (const float*)d_in[14];
    const float* cls_token  = (const float*)d_in[15];
    const float* ln_w       = (const float*)d_in[16];
    const float* ln_b       = (const float*)d_in[17];
    const float* rms_w      = (const float*)d_in[18];
    const float* bn_g       = (const float*)d_in[19];
    const float* bn_b       = (const float*)d_in[20];
    const float* gate       = (const float*)d_in[21];
    const float* q_w        = (const float*)d_in[22];
    const float* k_w        = (const float*)d_in[23];
    const float* v_w        = (const float*)d_in[24];
    const float* conv1_w    = (const float*)d_in[25];
    const float* conv1_b    = (const float*)d_in[26];
    const float* conv3_w    = (const float*)d_in[27];
    const float* conv3_b    = (const float*)d_in[28];
    const float* conv5_w    = (const float*)d_in[29];
    const float* conv5_b    = (const float*)d_in[30];
    const float* o_w        = (const float*)d_in[31];
    const float* o_b        = (const float*)d_in[32];
    const float* ffn_w1     = (const float*)d_in[33];
    const float* ffn_b1     = (const float*)d_in[34];
    const float* ffn_w2     = (const float*)d_in[35];
    const float* ffn_b2     = (const float*)d_in[36];
    const float* lm_w       = (const float*)d_in[37];
    const float* lm_b       = (const float*)d_in[38];
    const float* lv_w       = (const float*)d_in[39];
    const float* lv_b       = (const float*)d_in[40];
    (void)in_sizes; (void)n_in; (void)out_size; (void)ws_size;

    const int Bsz = 512, F = 64, L = 65, D = 256, Hn = 8, NL = 6;
    const int Mrows = Bsz * L;                  // 33280 = 260*128
    const int MF    = Bsz * F;                  // 32768

    char* ws = (char*)d_ws;
    size_t off = 0;
    auto alloc = [&](size_t bytes) { size_t o = off; off += (bytes + 255) & ~(size_t)255; return o; };

    float* hbuf   = (float*)(ws + alloc((size_t)Mrows * D * 4));
    h16*   q16    = (h16*)  (ws + alloc((size_t)Mrows * D * 2));
    h16*   k16    = (h16*)  (ws + alloc((size_t)Mrows * D * 2));
    float* vbuf   = (float*)(ws + alloc((size_t)Mrows * D * 4));
    float* vc     = (float*)(ws + alloc((size_t)Mrows * D * 4));      // also proj tmp
    float* attnP  = (float*)(ws + alloc((size_t)Bsz * Hn * L * L * 4)); // also act16 (M*1024 f16)
    h16*   Ah     = (h16*)  (ws + alloc((size_t)Mrows * 768 * 2));    // comb16 / hn16 / outs16
    h16*   W16    = (h16*)  (ws + alloc((size_t)14500000));
    float* edges  = (float*)(ws + alloc(64 * 34 * 4));
    float* xn     = (float*)(ws + alloc((size_t)MF * 4));
    float* odt    = (float*)(ws + alloc((size_t)Bsz * 64 * 4));
    h16*   act16  = (h16*)attnP;

    // convert + transpose: src f32 [K][N] -> dst f16 [N][K]
    auto CVT_T = [&](const float* s, h16* d, int K, int N) {
        int n = K * N;
        cvt_t_kernel<<<dim3((n + 255) / 256), 256, 0, stream>>>(s, d, K, N);
    };

    // --- weight conversion f32 -> f16 (transposed for WMMA B-side)
    const size_t LW = 3 * 65536 + 196608 + 524288 + 262144;   // per-layer halfs
    h16* qw16[NL]; h16* kw16[NL]; h16* vw16[NL]; h16* ow16[NL]; h16* f1w16[NL]; h16* f2w16[NL];
    for (int l = 0; l < NL; ++l) {
        h16* base = W16 + (size_t)l * LW;
        qw16[l] = base;            kw16[l] = base + 65536;    vw16[l] = base + 131072;
        ow16[l] = base + 196608;   f1w16[l] = base + 393216;  f2w16[l] = base + 917504;
        CVT_T(q_w  + (size_t)l * 65536,  qw16[l], 256, 256);
        CVT_T(k_w  + (size_t)l * 65536,  kw16[l], 256, 256);
        CVT_T(v_w  + (size_t)l * 65536,  vw16[l], 256, 256);
        CVT_T(o_w  + (size_t)l * 196608, ow16[l], 768, 256);
        CVT_T(ffn_w1 + (size_t)l * 524288, f1w16[l], 256, 2048);  // -> [2048][256]
        CVT_T(ffn_w2 + (size_t)l * 262144, f2w16[l], 1024, 256);  // -> [256][1024]
    }
    h16* projw16 = W16 + (size_t)NL * LW;     // [256][320]
    h16* lmw16   = projw16 + 81920;           // [128][256]
    h16* lvw16   = lmw16 + 32768;             // [128][256]
    CVT_T(proj_w, projw16, 320, 256);
    CVT_T(lm_w,   lmw16,   256, 128);
    CVT_T(lv_w,   lvw16,   256, 128);

    // --- preprocessing
    xn_kernel   <<<dim3(Bsz), 64, 0, stream>>>(x, fmean, fscale, xn);
    edges_kernel<<<dim3(64), 32, 0, stream>>>(bin_logits, edges);
    odt_kernel  <<<dim3(Bsz), 64, 0, stream>>>(xn, sel_w, sel_b, leaf_values, odt);
    tok_kernel  <<<dim3(MF), 128, 0, stream>>>(xn, fp_w1, fp_b1, fp_w2, fp_b2,
                                               pos_emb, edges, bin_emb_w, odt, Ah);
    // proj: tmp = comb16 @ proj_w + proj_b   (M=32768, K=320, N=256)
    gemm_wmma<0, false><<<dim3(D / BN, MF / BM), 256, 0, stream>>>(
        Ah, 320, projw16, 320, proj_b, vc, nullptr, 256, 320);
    assemble_kernel<<<dim3(Mrows), 256, 0, stream>>>(vc, cls_token, hbuf);

    // --- transformer layers
    dim3 gD(D / BN, Mrows / BM);
    for (int l = 0; l < NL; ++l) {
        anorm_kernel<<<dim3(Mrows), 256, 0, stream>>>(hbuf, 256, ln_w, ln_b, rms_w,
                                                      bn_g, bn_b, gate, 2 * l, Ah);
        gemm_wmma<1, false><<<gD, 256, 0, stream>>>(Ah, 256, qw16[l], 256, nullptr,
                                                    nullptr, q16, 256, 256);
        gemm_wmma<1, false><<<gD, 256, 0, stream>>>(Ah, 256, kw16[l], 256, nullptr,
                                                    nullptr, k16, 256, 256);
        gemm_wmma<0, false><<<gD, 256, 0, stream>>>(Ah, 256, vw16[l], 256, nullptr,
                                                    vbuf, nullptr, 256, 256);
        attn_kernel<<<dim3(Bsz * Hn), 128, 0, stream>>>(q16, k16, attnP);

        const float* cws[3] = { conv1_w + (size_t)l * 256 * 32,
                                conv3_w + (size_t)l * 256 * 32 * 3,
                                conv5_w + (size_t)l * 256 * 32 * 5 };
        const float* cbs[3] = { conv1_b + (size_t)l * 256,
                                conv3_b + (size_t)l * 256,
                                conv5_b + (size_t)l * 256 };
        const int kss[3] = { 1, 3, 5 };
        for (int s = 0; s < 3; ++s) {
            gconv_kernel<<<dim3(Bsz), 256, 0, stream>>>(vbuf, cws[s], cbs[s], kss[s], vc);
            av_kernel<<<dim3(Bsz), 256, 0, stream>>>(attnP, vc, Ah, s * 256);
        }
        // h += outs16 @ o_w + o_b   (K=768)
        gemm_wmma<0, true><<<gD, 256, 0, stream>>>(Ah, 768, ow16[l], 768,
                                                   o_b + (size_t)l * 256,
                                                   hbuf, nullptr, 256, 768);
        anorm_kernel<<<dim3(Mrows), 256, 0, stream>>>(hbuf, 256, ln_w, ln_b, rms_w,
                                                      bn_g, bn_b, gate, 2 * l + 1, Ah);
        // act16 = gelu(hn2 @ Wu + bu) * (hn2 @ Wg + bg)   (N=1024)
        gemm_ffn1<<<dim3(1024 / BN, Mrows / BM), 256, 0, stream>>>(
            Ah, 256, f1w16[l], f1w16[l] + (size_t)1024 * 256, 256,
            ffn_b1 + (size_t)l * 2048, act16, 1024, 256);
        // h += act16 @ ffn_w2 + ffn_b2   (K=1024)
        gemm_wmma<0, true><<<gD, 256, 0, stream>>>(act16, 1024, f2w16[l], 1024,
                                                   ffn_b2 + (size_t)l * 256,
                                                   hbuf, nullptr, 256, 1024);
    }

    // --- final norm on CLS rows (gathered with stride 65*256) + heads
    h16* cls16 = Ah;                                     // 512 x 256 f16
    anorm_kernel<<<dim3(Bsz), 256, 0, stream>>>(hbuf, (long long)L * 256, ln_w, ln_b,
                                                rms_w, bn_g, bn_b, gate, 12, cls16);
    float* out = (float*)d_out;                          // (512, 256) = [mu | logvar]
    gemm_wmma<0, false><<<dim3(128 / BN, Bsz / BM), 256, 0, stream>>>(
        cls16, 256, lmw16, 256, lm_b, out, nullptr, 256, 256);
    gemm_wmma<0, false><<<dim3(128 / BN, Bsz / BM), 256, 0, stream>>>(
        cls16, 256, lvw16, 256, lv_b, out + 128, nullptr, 256, 256);
}
